// TransformerSpotV2Net_84207128805742
// MI455X (gfx1250) — matrix operand order repaired and checked by reading, not compiled, not run on hardware
//
#include <hip/hip_runtime.h>
#include <math.h>

// ---------------------------------------------------------------------------
// TransformerConv GNN forward for MI455X (gfx1250, wave32).
//
//  * Dense GEMMs run on v_wmma_f32_16x16x32_f16 (f32 accum). A-panel (16xK
//    per block) is DMA'd into LDS by the Tensor Data Mover (tensor_load_to_lds
//    + s_wait_tensorcnt); B tiles are cooperatively loaded and stored
//    TRANSPOSED in LDS so both A and B fragments are contiguous ds_load_b128.
//  * edge_attr is scalar -> per-layer edge embedding collapses to s_e*u + w0
//    (u = Wep@We, w0 = bep@We), removing the [E,32]@[32,256] GEMM.
//  * Edge phase: one wave per edge, 8 ch/lane; segment softmax via
//    order-preserving-uint atomicMax + global_atomic_add_f32. Node state
//    (30MB/tensor) is L2-resident (192MB), so gathers/scatters stay on-chip.
//  * Workspace: ~325 MB carve below.
//
// d_in index map (top-level dict insertion order; params tree-flattened with
// sorted keys, lists in order):
//   0 x 1 edge_index(i32) 2 edge_attr 3 Wep 4 Win 5 bep 6 bin
//   7..18  heads[j]: {W1,W2,b1,b2} stride 4
//   19..34 layers[0]: W1,W2,Wb,We,Wk,Wq,Ws,Wv,b1,b2,bk,bq,bs,bv,ln_b,ln_g
//   35..50 layers[1] same; 51..62 layers[2]: Wb,We,Wk,Wq,Ws,Wv,bk,bq,bs,bv,ln_b,ln_g
//   63 out_b 64 out_g
// ---------------------------------------------------------------------------

#define NNODES 30000
#define NEDGES 480000
#define HDIM   256
#define KMAX   512

typedef _Float16 half_t;
typedef __attribute__((ext_vector_type(16))) _Float16 v16h;
typedef __attribute__((ext_vector_type(8)))  float    v8f;
typedef __attribute__((ext_vector_type(4)))  unsigned u32x4;
typedef __attribute__((ext_vector_type(8)))  int      i32x8;
typedef __attribute__((ext_vector_type(4)))  int      i32x4;

union V16 { v16h v; uint4 u4[2]; };

static __device__ __forceinline__ float gelu_exact(float x) {
  return 0.5f * x * (1.0f + erff(x * 0.7071067811865475f));
}
static __device__ __forceinline__ unsigned fenc(float f) {
  unsigned u = __float_as_uint(f);
  return (u & 0x80000000u) ? ~u : (u | 0x80000000u);
}
static __device__ __forceinline__ float fdec(unsigned u) {
  return (u & 0x80000000u) ? __uint_as_float(u & 0x7FFFFFFFu)
                           : __uint_as_float(~u);
}

// ---------------------------------------------------------------------------
// TDM: 2D tile load Global->LDS. D# packing per CDNA5 ISA ch.8:
//  group0: [1:0]count=1 | [63:32]lds_addr | [120:64]global_addr | [127:126]type=2
//  group1: [17:16]data_size=1(2B) | [79:48]dim0 | [111:80]dim1 | [127:112]tile0
//          [143:128]tile1 | [207:160]dim0_stride
// Tensor sized exactly = tile (no OOB clipping needed; all our tiles are full).
// ---------------------------------------------------------------------------
static __device__ __forceinline__ void tdm_load_2d(unsigned lds_off,
                                                   const void* gptr,
                                                   unsigned dim0_elems,
                                                   unsigned dim1_rows,
                                                   unsigned stride_elems) {
  unsigned long long ga = (unsigned long long)gptr;
  u32x4 g0;
  g0.x = 1u;                                   // count=1, user mode
  g0.y = lds_off;                              // LDS byte address
  g0.z = (unsigned)(ga & 0xFFFFFFFFu);         // global_addr[31:0]
  g0.w = (unsigned)((ga >> 32) & 0x01FFFFFFu)  // global_addr[56:32]
       | 0x80000000u;                          // type=2 ("image")
  i32x8 g1 = {};
  g1[0] = 0x00010000;                          // data_size=1 -> 2 bytes/elem
  g1[1] = (int)(dim0_elems << 16);             // tensor_dim0[15:0]
  g1[2] = (int)((dim0_elems >> 16) | (dim1_rows << 16));
  g1[3] = (int)((dim1_rows >> 16) | (dim0_elems << 16)); // tile_dim0 = dim0
  g1[4] = (int)(dim1_rows & 0xFFFF);           // tile_dim1 = dim1 (tile2=0)
  g1[5] = (int)stride_elems;                   // tensor_dim0_stride[31:0]
  i32x4 z4 = {};
#if __clang_major__ >= 23
  i32x8 z8 = {};
  __builtin_amdgcn_tensor_load_to_lds(g0, g1, z4, z4, z8, 0);
#else
  __builtin_amdgcn_tensor_load_to_lds(g0, g1, z4, z4, 0);
#endif
}

// ---------------------------------------------------------------------------
// WMMA GEMM: out[M,Dout] = act(A16[M,K] @ W16[K,Dout] + bias).
// M%16==0, 32<=K<=KMAX, K%32==0, Dout%64==0. Block = 128 threads (4 waves),
// 16x64 output tile. A panel (16xK) arrives via TDM; W tiles (32x64) are
// cooperatively loaded and stored transposed so fragments are ds_load_b128.
// ---------------------------------------------------------------------------
__global__ __launch_bounds__(128)
void gemm_wmma_kernel(const half_t* __restrict__ A, const half_t* __restrict__ W,
                      const float* __restrict__ bias, float* __restrict__ out,
                      int M, int K, int Dout, int act) {
  __shared__ half_t lA[16 * KMAX];   // 16 KB: full A panel for this block row
  __shared__ half_t lBt[64 * 32];    // 4 KB: transposed B tile [col][k]
  const int t    = threadIdx.x;
  const int lane = t & 31;
  const int wave = t >> 5;
  const int row0 = blockIdx.x * 16;
  const int colB = blockIdx.y * 64;
  const int col  = colB + wave * 16 + (lane & 15);
  const int hiA  = (lane >> 4) << 3;   // A half-chunk base within 32-K slice
  const int hiB  = (lane >> 4) << 3;   // B K-row base: 0 or 8

  // --- Tensor Data Mover: pull the 16xK A panel into LDS (wave 0 only) ---
  if (wave == 0) {
    tdm_load_2d((unsigned)(size_t)&lA[0], A + (size_t)row0 * K,
                (unsigned)K, 16u, (unsigned)K);
    __builtin_amdgcn_s_wait_tensorcnt(0);
  }
  __syncthreads();

  v8f acc = {};
  for (int kk = 0; kk < K; kk += 32) {
    // cooperative W tile load (coalesced) + transposed LDS store
    {
      int r = t >> 2, c0 = (t & 3) * 16;
      const half_t* gp = W + (size_t)(kk + r) * Dout + colB + c0;
      half_t tmp[16];
      *(uint4*)&tmp[0] = *(const uint4*)gp;
      *(uint4*)&tmp[8] = *(const uint4*)(gp + 8);
      __syncthreads();                    // protect prior-iter fragment reads
#pragma unroll
      for (int j = 0; j < 16; ++j) lBt[(c0 + j) * 32 + r] = tmp[j];
    }
    __syncthreads();
    // A fragment (ISA 7.12.2): row = lane&15; halves 0..7 -> K=hiA..hiA+7,
    // halves 8..15 -> K=hiA+16..hiA+23  (two contiguous b128 loads)
    V16 af, bf;
    const half_t* ap = &lA[(lane & 15) * K + kk + hiA];
    af.u4[0] = *(const uint4*)ap;
    af.u4[1] = *(const uint4*)(ap + 16);
    // B fragment: col = lane&15 of wave tile, K rows hiB..hiB+7 and hiB+16..hiB+23 (as A)
    const half_t* bp = &lBt[(wave * 16 + (lane & 15)) * 32 + hiB];
    bf.u4[0] = *(const uint4*)bp;
    bf.u4[1] = *(const uint4*)(bp + 16);
    acc = __builtin_amdgcn_wmma_f32_16x16x32_f16(
        false, af.v, false, bf.v, (short)0, acc, false, false);
  }
  // C layout: VGPR r -> (M = row0 + r + 8*(lane>>4), N = col)
  const float bv = bias[col];
  float* op = out + (size_t)(row0 + ((lane >> 4) << 3)) * Dout + col;
#pragma unroll
  for (int r = 0; r < 8; ++r) {
    float y = acc[r] + bv;
    op[(size_t)r * Dout] = act ? gelu_exact(y) : y;
  }
}

// ---------------------------------------------------------------------------
// small elementwise / init kernels
// ---------------------------------------------------------------------------
__global__ void cvt_f32_f16_kernel(const float* __restrict__ s,
                                   half_t* __restrict__ d, size_t n) {
  for (size_t i = blockIdx.x * (size_t)blockDim.x + threadIdx.x; i < n;
       i += (size_t)gridDim.x * blockDim.x)
    d[i] = (half_t)s[i];
}
__global__ void fill_f32_kernel(float* p, float v, size_t n) {
  for (size_t i = blockIdx.x * (size_t)blockDim.x + threadIdx.x; i < n;
       i += (size_t)gridDim.x * blockDim.x)
    p[i] = v;
}
__global__ void fill_u32_kernel(unsigned* p, unsigned v, size_t n) {
  for (size_t i = blockIdx.x * (size_t)blockDim.x + threadIdx.x; i < n;
       i += (size_t)gridDim.x * blockDim.x)
    p[i] = v;
}
__global__ void add_kernel(float* __restrict__ o, const float* __restrict__ a,
                           const float* __restrict__ b, size_t n) {
  for (size_t i = blockIdx.x * (size_t)blockDim.x + threadIdx.x; i < n;
       i += (size_t)gridDim.x * blockDim.x)
    o[i] = a[i] + b[i];
}

// input projection: h = x[N,31] @ Win[31,256] + bin. One wave per node.
__global__ void input_proj_kernel(const float* __restrict__ x,
                                  const float* __restrict__ Win,
                                  const float* __restrict__ bin,
                                  float* __restrict__ h, int N) {
  int wave = threadIdx.x >> 5, lane = threadIdx.x & 31;
  int n = blockIdx.x * (blockDim.x >> 5) + wave;
  if (n >= N) return;
#pragma unroll
  for (int i = 0; i < 8; ++i) {
    int f = lane * 8 + i;
    float acc = bin[f];
    for (int c = 0; c < 31; ++c) acc += x[(size_t)n * 31 + c] * Win[c * HDIM + f];
    h[(size_t)n * HDIM + f] = acc;
  }
}

// u = Wep @ We, w0 = bep @ We  (per layer; 256 threads)
__global__ void edge_uv_kernel(const float* __restrict__ Wep,
                               const float* __restrict__ bep,
                               const float* __restrict__ We,
                               float* __restrict__ uv) {
  int j = threadIdx.x;
  float u = 0.f, w = 0.f;
  for (int c = 0; c < 32; ++c) {
    float wv = We[c * HDIM + j];
    u += Wep[c] * wv;
    w += bep[c] * wv;
  }
  uv[j] = u;
  uv[HDIM + j] = w;
}

// pass A: alpha[e,h] = <q[dst], k[src] + s*u + w0>_h / sqrt(C); atomicMax amax
__global__ void edge_alpha_kernel(const int* __restrict__ eidx,
                                  const float* __restrict__ eattr,
                                  const float* __restrict__ q,
                                  const float* __restrict__ k,
                                  const float* __restrict__ uv,
                                  float* __restrict__ alpha,
                                  unsigned* __restrict__ amax, int E, int heads) {
  int wave = threadIdx.x >> 5, lane = threadIdx.x & 31;
  int e = blockIdx.x * (blockDim.x >> 5) + wave;
  if (e >= E) return;
  int src = eidx[e], dst = eidx[E + e];
  float s = eattr[e];
  const float* u = uv; const float* w0 = uv + HDIM;
  int f0 = lane * 8;
  float p = 0.f;
#pragma unroll
  for (int i = 0; i < 8; ++i) {
    int f = f0 + i;
    float kj = k[(size_t)src * HDIM + f] + s * u[f] + w0[f];
    p += q[(size_t)dst * HDIM + f] * kj;
  }
  int lph = 32 / heads;                       // lanes per head (4 or 32)
  for (int m = lph >> 1; m >= 1; m >>= 1) p += __shfl_xor(p, m);
  float a = p * rsqrtf((float)(HDIM / heads));
  if ((lane & (lph - 1)) == 0) {
    int h = lane / lph;
    alpha[(size_t)e * heads + h] = a;
    atomicMax(&amax[(size_t)dst * heads + h], fenc(a));
  }
}

// pass B: ex = exp(alpha - amax[dst]); scatter-add denominator + numerator
__global__ void edge_scatter_kernel(const int* __restrict__ eidx,
                                    const float* __restrict__ eattr,
                                    const float* __restrict__ v,
                                    const float* __restrict__ uv,
                                    const float* __restrict__ alpha,
                                    const unsigned* __restrict__ amax,
                                    float* __restrict__ S, float* __restrict__ D,
                                    int E, int heads) {
  int wave = threadIdx.x >> 5, lane = threadIdx.x & 31;
  int e = blockIdx.x * (blockDim.x >> 5) + wave;
  if (e >= E) return;
  int src = eidx[e], dst = eidx[E + e];
  float s = eattr[e];
  const float* u = uv; const float* w0 = uv + HDIM;
  int lph = 32 / heads;
  int h = lane / lph;
  float am = fdec(amax[(size_t)dst * heads + h]);
  float ex = expf(alpha[(size_t)e * heads + h] - am);
  if ((lane & (lph - 1)) == 0) atomicAdd(&D[(size_t)dst * heads + h], ex);
  int f0 = lane * 8;
#pragma unroll
  for (int i = 0; i < 8; ++i) {
    int f = f0 + i;
    float m = ex * (v[(size_t)src * HDIM + f] + s * u[f] + w0[f]);
    atomicAdd(&S[(size_t)dst * HDIM + f], m);
  }
}

// node combine: out = S/(D+1e-16); beta-gated skip; LayerNorm; GELU -> h
__global__ void node_combine_kernel(const float* __restrict__ S,
                                    const float* __restrict__ D,
                                    const float* __restrict__ xr,
                                    const float* __restrict__ Wb,
                                    const float* __restrict__ lng,
                                    const float* __restrict__ lnb,
                                    float* __restrict__ hout, int N, int heads) {
  int wave = threadIdx.x >> 5, lane = threadIdx.x & 31;
  int n = blockIdx.x * (blockDim.x >> 5) + wave;
  if (n >= N) return;
  int lph = 32 / heads;
  float dden = D[(size_t)n * heads + lane / lph] + 1e-16f;
  int f0 = lane * 8;
  float o[8], xv[8];
  float bdot = 0.f;
#pragma unroll
  for (int i = 0; i < 8; ++i) {
    int f = f0 + i;
    o[i]  = S[(size_t)n * HDIM + f] / dden;
    xv[i] = xr[(size_t)n * HDIM + f];
    bdot += o[i] * Wb[f] + xv[i] * Wb[HDIM + f] + (o[i] - xv[i]) * Wb[2 * HDIM + f];
  }
  for (int m = 16; m >= 1; m >>= 1) bdot += __shfl_xor(bdot, m);
  float beta = 1.0f / (1.0f + expf(-bdot));
  float hv[8], ssum = 0.f;
#pragma unroll
  for (int i = 0; i < 8; ++i) { hv[i] = beta * xv[i] + (1.0f - beta) * o[i]; ssum += hv[i]; }
  for (int m = 16; m >= 1; m >>= 1) ssum += __shfl_xor(ssum, m);
  float mean = ssum * (1.0f / HDIM);
  float vsum = 0.f;
#pragma unroll
  for (int i = 0; i < 8; ++i) { float d = hv[i] - mean; vsum += d * d; }
  for (int m = 16; m >= 1; m >>= 1) vsum += __shfl_xor(vsum, m);
  float inv = rsqrtf(vsum * (1.0f / HDIM) + 1e-5f);
#pragma unroll
  for (int i = 0; i < 8; ++i) {
    int f = f0 + i;
    float y = (hv[i] - mean) * inv * lng[f] + lnb[f];
    hout[(size_t)n * HDIM + f] = gelu_exact(y);
  }
}

// plain LayerNorm (final), no activation
__global__ void ln_kernel(const float* __restrict__ in, const float* __restrict__ g,
                          const float* __restrict__ b, float* __restrict__ out, int N) {
  int wave = threadIdx.x >> 5, lane = threadIdx.x & 31;
  int n = blockIdx.x * (blockDim.x >> 5) + wave;
  if (n >= N) return;
  int f0 = lane * 8;
  float hv[8], ssum = 0.f;
#pragma unroll
  for (int i = 0; i < 8; ++i) { hv[i] = in[(size_t)n * HDIM + f0 + i]; ssum += hv[i]; }
  for (int m = 16; m >= 1; m >>= 1) ssum += __shfl_xor(ssum, m);
  float mean = ssum * (1.0f / HDIM);
  float vsum = 0.f;
#pragma unroll
  for (int i = 0; i < 8; ++i) { float d = hv[i] - mean; vsum += d * d; }
  for (int m = 16; m >= 1; m >>= 1) vsum += __shfl_xor(vsum, m);
  float inv = rsqrtf(vsum * (1.0f / HDIM) + 1e-5f);
#pragma unroll
  for (int i = 0; i < 8; ++i) {
    int f = f0 + i;
    out[(size_t)n * HDIM + f] = (hv[i] - mean) * inv * g[f] + b[f];
  }
}

// output head finalize: out[n] += (t[n,:64] . W2 + b2)/3   (t already GELU'd)
__global__ void head_out_kernel(const float* __restrict__ t,
                                const float* __restrict__ W2,
                                const float* __restrict__ b2,
                                float* __restrict__ out, int N) {
  int wave = threadIdx.x >> 5, lane = threadIdx.x & 31;
  int n = blockIdx.x * (blockDim.x >> 5) + wave;
  if (n >= N) return;
  float p = 0.f;
#pragma unroll
  for (int i = 0; i < 2; ++i) {
    int u = lane * 2 + i;
    p += t[(size_t)n * 64 + u] * W2[u];
  }
  for (int m = 16; m >= 1; m >>= 1) p += __shfl_xor(p, m);
  if (lane == 0) out[n] += (p + b2[0]) * (1.0f / 3.0f);
}

// ---------------------------------------------------------------------------
extern "C" void kernel_launch(void* const* d_in, const int* in_sizes, int n_in,
                              void* d_out, int out_size, void* d_ws, size_t ws_size,
                              hipStream_t stream) {
  (void)in_sizes; (void)n_in; (void)out_size; (void)ws_size;
  const int N = NNODES, E = NEDGES;
  const float* x     = (const float*)d_in[0];
  const int*   eidx  = (const int*)  d_in[1];
  const float* eattr = (const float*)d_in[2];
  const float* Wep = (const float*)d_in[3];
  const float* Win = (const float*)d_in[4];
  const float* bep = (const float*)d_in[5];
  const float* bin = (const float*)d_in[6];
  const int IDX_HEADS = 7;                       // +4j: W1,W2,b1,b2
  const int LBASE[3] = {19, 35, 51};
  auto Lp = [&](int L, int off) { return (const float*)d_in[LBASE[L] + off]; };
  const float* out_b = (const float*)d_in[63];
  const float* out_g = (const float*)d_in[64];

  // ---- workspace carve (~325 MB) ----
  char* wp = (char*)d_ws;
  auto carve = [&](size_t bytes) { char* p = wp; wp += (bytes + 255) & ~(size_t)255; return p; };
  const size_t FH = (size_t)N * HDIM * sizeof(float);
  float*  h     = (float*) carve(FH);
  float*  hprev = (float*) carve(FH);
  float*  q     = (float*) carve(FH);
  float*  k     = (float*) carve(FH);
  float*  v     = (float*) carve(FH);
  float*  xr    = (float*) carve(FH);
  float*  S     = (float*) carve(FH);
  float*  D     = (float*) carve((size_t)N * 8 * sizeof(float));
  unsigned* amax= (unsigned*)carve((size_t)N * 8 * sizeof(unsigned));
  float*  alpha = (float*) carve((size_t)E * 8 * sizeof(float));
  float*  uv    = (float*) carve(2 * HDIM * sizeof(float));
  float*  tmpff = (float*) carve((size_t)N * 512 * sizeof(float));
  half_t* a16   = (half_t*)carve((size_t)N * 512 * sizeof(half_t));
  half_t* w16   = (half_t*)carve((size_t)512 * 256 * sizeof(half_t));

  const int EB = (E + 7) / 8;
  const int NB = (N + 7) / 8;
  auto gsz = [](size_t n) { size_t b = (n + 255) / 256; return (int)(b > 8192 ? 8192 : b); };
  auto cvt = [&](const float* src, half_t* dst, size_t n) {
    cvt_f32_f16_kernel<<<gsz(n), 256, 0, stream>>>(src, dst, n);
  };
  auto gemm = [&](const half_t* A, const float* Wf, const float* bias,
                  float* out, int K, int Dout, int act) {
    cvt(Wf, w16, (size_t)K * Dout);
    dim3 g(N / 16, Dout / 64);
    gemm_wmma_kernel<<<g, 128, 0, stream>>>(A, w16, bias, out, N, K, Dout, act);
  };

  // ---- input projection ----
  input_proj_kernel<<<NB, 256, 0, stream>>>(x, Win, bin, h, N);

  // ---- 3 TransformerConv layers ----
  for (int L = 0; L < 3; ++L) {
    const int heads = (L < 2) ? 8 : 1;
    const bool ff = (L < 2);
    const float *W1, *W2, *Wb, *We, *Wk, *Wq, *Ws, *Wv;
    const float *b1, *b2, *bk, *bq, *bs, *bv, *lnb, *lng;
    if (ff) {
      W1 = Lp(L,0);  W2 = Lp(L,1);  Wb = Lp(L,2);  We = Lp(L,3);
      Wk = Lp(L,4);  Wq = Lp(L,5);  Ws = Lp(L,6);  Wv = Lp(L,7);
      b1 = Lp(L,8);  b2 = Lp(L,9);  bk = Lp(L,10); bq = Lp(L,11);
      bs = Lp(L,12); bv = Lp(L,13); lnb= Lp(L,14); lng= Lp(L,15);
    } else {
      W1 = W2 = b1 = b2 = nullptr;
      Wb = Lp(L,0);  We = Lp(L,1);  Wk = Lp(L,2);  Wq = Lp(L,3);
      Ws = Lp(L,4);  Wv = Lp(L,5);  bk = Lp(L,6);  bq = Lp(L,7);
      bs = Lp(L,8);  bv = Lp(L,9);  lnb= Lp(L,10); lng= Lp(L,11);
    }
    hipMemcpyAsync(hprev, h, FH, hipMemcpyDeviceToDevice, stream);

    // Q/K/V/skip projections via WMMA (+TDM-staged A panel)
    cvt(h, a16, (size_t)N * HDIM);
    gemm(a16, Wq, bq, q,  HDIM, HDIM, 0);
    gemm(a16, Wk, bk, k,  HDIM, HDIM, 0);
    gemm(a16, Wv, bv, v,  HDIM, HDIM, 0);
    gemm(a16, Ws, bs, xr, HDIM, HDIM, 0);

    // edge-embedding collapse + segment softmax + scatter
    edge_uv_kernel<<<1, 256, 0, stream>>>(Wep, bep, We, uv);
    fill_u32_kernel<<<gsz((size_t)N * heads), 256, 0, stream>>>(amax, 0u, (size_t)N * heads);
    fill_f32_kernel<<<gsz((size_t)N * heads), 256, 0, stream>>>(D, 0.f, (size_t)N * heads);
    fill_f32_kernel<<<gsz((size_t)N * HDIM), 256, 0, stream>>>(S, 0.f, (size_t)N * HDIM);
    edge_alpha_kernel  <<<EB, 256, 0, stream>>>(eidx, eattr, q, k, uv, alpha, amax, E, heads);
    edge_scatter_kernel<<<EB, 256, 0, stream>>>(eidx, eattr, v, uv, alpha, amax, S, D, E, heads);
    node_combine_kernel<<<NB, 256, 0, stream>>>(S, D, xr, Wb, lng, lnb, h, N, heads);

    if (ff) {  // h = gelu(h@W1+b1)@W2 + b2 + hprev   (GELU fused in GEMM1)
      cvt(h, a16, (size_t)N * HDIM);
      gemm(a16, W1, b1, tmpff, HDIM, 512, 1);
      cvt(tmpff, a16, (size_t)N * 512);
      gemm(a16, W2, b2, q, 512, HDIM, 0);     // q as scratch
      add_kernel<<<gsz((size_t)N * HDIM), 256, 0, stream>>>(h, q, hprev, (size_t)N * HDIM);
    }
  }

  // ---- final LN + 3 output heads, averaged ----
  ln_kernel<<<NB, 256, 0, stream>>>(h, out_g, out_b, S, N);   // S := normalized h
  fill_f32_kernel<<<gsz(N), 256, 0, stream>>>((float*)d_out, 0.f, (size_t)N);
  cvt(S, a16, (size_t)N * HDIM);
  for (int j = 0; j < 3; ++j) {
    const float* hW1 = (const float*)d_in[IDX_HEADS + 4 * j + 0];
    const float* hW2 = (const float*)d_in[IDX_HEADS + 4 * j + 1];
    const float* hb1 = (const float*)d_in[IDX_HEADS + 4 * j + 2];
    const float* hb2 = (const float*)d_in[IDX_HEADS + 4 * j + 3];
    gemm(a16, hW1, hb1, tmpff, HDIM, 64, 1);   // GELU fused
    head_out_kernel<<<NB, 256, 0, stream>>>(tmpff, hW2, hb2, (float*)d_out, N);
  }
}
